// SelfMultiheadAttn_43121471652290
// MI455X (gfx1250) — matrix operand order repaired and checked by reading, not compiled, not run on hardware
//
#include <hip/hip_runtime.h>
#include <hip/hip_bf16.h>
#include <math.h>
#include <stdint.h>

// ---------------------------------------------------------------------------
// Types for CDNA5 WMMA (wave32): v_wmma_f32_16x16x32_bf16
// ---------------------------------------------------------------------------
typedef __bf16 v16bf __attribute__((ext_vector_type(16)));
typedef __bf16 v8bf  __attribute__((ext_vector_type(8)));
typedef __bf16 v4bf  __attribute__((ext_vector_type(4)));
typedef float  v8f   __attribute__((ext_vector_type(8)));

union Frag {
    v16bf v;
    v8bf  h[2];
};

static constexpr int T     = 1024;
static constexpr int Bsz   = 8;
static constexpr int E     = 1024;
static constexpr int H     = 16;
static constexpr int D     = 64;     // E / H
static constexpr int M     = T * Bsz;   // 8192 rows for the projections
static constexpr int KDIM  = E;         // 1024 reduction dim for both GEMMs
static constexpr int N_QKV = 3 * E;     // 3072

// ---------------------------------------------------------------------------
// Optional CDNA5 async Global->LDS path (guarded; falls back cleanly).
// The builtin takes (global int4* src, local int4* dst, imm offset, imm cpol)
// per the toolchain's parameter diagnostic.
// ---------------------------------------------------------------------------
#if defined(__has_builtin)
#if __has_builtin(__builtin_amdgcn_global_load_async_to_lds_b128) && \
    __has_builtin(__builtin_amdgcn_s_wait_asynccnt)
#define HAVE_ASYNC_LDS 1
#endif
#endif
#ifndef HAVE_ASYNC_LDS
#define HAVE_ASYNC_LDS 0
#endif

#if HAVE_ASYNC_LDS
typedef int v4i __attribute__((ext_vector_type(4)));
typedef __attribute__((address_space(1))) v4i g_v4i;
typedef __attribute__((address_space(3))) v4i l_v4i;
// Global generic pointer value == AS1 address.
__device__ __forceinline__ g_v4i* to_gbl_v4(const void* p) {
    return (g_v4i*)(uintptr_t)p;
}
// Generic LDS address: low 32 bits are the LDS byte offset (ISA aperture rule).
__device__ __forceinline__ l_v4i* to_lds_v4(void* p) {
    return (l_v4i*)(uint32_t)(uintptr_t)p;
}
#endif

__device__ __forceinline__ __bf16 f2bf(float f) {
    union { float f; unsigned int u; } in;
    in.f = f;
    unsigned int u = in.u;
    unsigned int r = (u + 0x7FFFu + ((u >> 16) & 1u)) >> 16;  // round-nearest-even
    union { unsigned short s; __bf16 b; } out;
    out.s = (unsigned short)r;
    return out.b;
}

__device__ __forceinline__ v8f vzero8() {
    v8f x;
#pragma unroll
    for (int i = 0; i < 8; ++i) x[i] = 0.0f;
    return x;
}

// ---------------------------------------------------------------------------
// fp32 -> bf16 conversion (vectorized by 4)
// ---------------------------------------------------------------------------
__global__ __launch_bounds__(256) void cvt_f32_to_bf16(const float* __restrict__ src,
                                                       __bf16* __restrict__ dst, int n4) {
    int i = blockIdx.x * 256 + threadIdx.x;
    if (i < n4) {
        const float4 v = reinterpret_cast<const float4*>(src)[i];
        v4bf o;
        o[0] = f2bf(v.x); o[1] = f2bf(v.y); o[2] = f2bf(v.z); o[3] = f2bf(v.w);
        reinterpret_cast<v4bf*>(dst)[i] = o;
    }
}

// ---------------------------------------------------------------------------
// Wave-level GEMM main loop: C(64x32 per wave) = A(row-major MxK) * W^T
// (W row-major NxK, so B-fragment column n == row n of W, K-contiguous).
// Workgroup = 8 waves arranged 2(M) x 4(N) -> 128x128 tile.
// ---------------------------------------------------------------------------
__device__ __forceinline__ void mm_wave_tile(const __bf16* __restrict__ A,
                                             const __bf16* __restrict__ W,
                                             int mBase, int nBase, v8f (&acc)[4][2]) {
    const int lane = threadIdx.x & 31;
    const int r    = lane & 15;
    const int kh   = lane >> 4;
    const __bf16* aRow[4];
    const __bf16* bRow[2];
#pragma unroll
    for (int i = 0; i < 4; ++i) aRow[i] = A + (size_t)(mBase + i * 16 + r) * KDIM;
#pragma unroll
    for (int j = 0; j < 2; ++j) bRow[j] = W + (size_t)(nBase + j * 16 + r) * KDIM;
#pragma unroll
    for (int i = 0; i < 4; ++i)
#pragma unroll
        for (int j = 0; j < 2; ++j) acc[i][j] = vzero8();

    for (int kk = 0; kk < KDIM; kk += 32) {
        Frag a[4], b[2];
#pragma unroll
        for (int i = 0; i < 4; ++i) {
            a[i].h[0] = *reinterpret_cast<const v8bf*>(aRow[i] + kk + kh * 8);
            a[i].h[1] = *reinterpret_cast<const v8bf*>(aRow[i] + kk + 16 + kh * 8);
        }
#pragma unroll
        for (int j = 0; j < 2; ++j) {
            const __bf16* p = bRow[j] + kk + kh * 8;
            b[j].h[0] = *reinterpret_cast<const v8bf*>(p);
            b[j].h[1] = *reinterpret_cast<const v8bf*>(p + 16);
        }
#pragma unroll
        for (int i = 0; i < 4; ++i)
#pragma unroll
            for (int j = 0; j < 2; ++j)
                acc[i][j] = __builtin_amdgcn_wmma_f32_16x16x32_bf16(
                    false, a[i].v, false, b[j].v, (short)0, acc[i][j], false, false);
    }
}

// ---------------------------------------------------------------------------
// QKV projection: P = X @ W1^T (M=8192, N=3072, K=1024).
// Epilogue: f = h*192 + c*64 + d (apex interleaved layout); q scaled by 1/8;
// scatter into per-head (B*H, T, D) Q/K/V bf16 buffers.
// ---------------------------------------------------------------------------
__global__ __launch_bounds__(256) void gemm_qkv(const __bf16* __restrict__ X,
                                                const __bf16* __restrict__ W1,
                                                __bf16* __restrict__ Qb,
                                                __bf16* __restrict__ Kb,
                                                __bf16* __restrict__ Vb) {
    const int w    = threadIdx.x >> 5;
    const int lane = threadIdx.x & 31;
    const int wm = w & 1, wn = w >> 1;
    const int r = lane & 15, kh = lane >> 4;
    const int mBase = blockIdx.y * 128 + wm * 64;
    const int nBase = blockIdx.x * 128 + wn * 32;

    v8f acc[4][2];
    mm_wave_tile(X, W1, mBase, nBase, acc);

#pragma unroll
    for (int i = 0; i < 4; ++i)
#pragma unroll
        for (int j = 0; j < 2; ++j)
#pragma unroll
            for (int e = 0; e < 8; ++e) {
                int m = mBase + i * 16 + e + 8 * kh;
                int f = nBase + j * 16 + r;
                int t = m >> 3, bb = m & 7;               // m = t*B + b
                int hh = f / 192;
                int rem = f - hh * 192;
                int c = rem >> 6;                          // 0=q, 1=k, 2=v
                int d = rem & 63;
                float val = acc[i][j][e];
                if (c == 0) val *= 0.125f;                 // D^-0.5
                __bf16* dp = (c == 0) ? Qb : ((c == 1) ? Kb : Vb);
                dp[(((size_t)(bb * H + hh)) * T + t) * D + d] = f2bf(val);
            }
}

// ---------------------------------------------------------------------------
// Flash attention: one workgroup per (head n, 128-query block).
// 8 waves; wave w owns 16 query rows. Online softmax, ctx accum in f32.
// K tile staged once per workgroup via the CDNA5 async Global->LDS engine
// (when available) -- removes the 8x redundant per-wave K reads.
// V tile staged transposed in LDS (pad 136 vs 128 to spread banks);
// P repacked to per-wave LDS for the P@V A-fragments.
// ---------------------------------------------------------------------------
__global__ __launch_bounds__(256) void flash_attn(const __bf16* __restrict__ Q,
                                                  const __bf16* __restrict__ Kg,
                                                  const __bf16* __restrict__ Vg,
                                                  __bf16* __restrict__ Ctx) {
    __shared__ __bf16 vT[64][136];        // V^T tile: [d][key]
    __shared__ __bf16 pS[8][16][136];     // per-wave probs: [wave][row][key]
#if HAVE_ASYNC_LDS
    __shared__ __bf16 kT[128][80];        // K tile row-major, padded (160B rows)
#endif

    const int n  = blockIdx.x >> 3;       // head index = b*H + h
    const int qb = blockIdx.x & 7;        // query block (T/128 = 8)
    const int w = threadIdx.x >> 5;
    const int lane = threadIdx.x & 31;
    const int r = lane & 15, kh = lane >> 4;

    const __bf16* Qh = Q  + (size_t)n * T * D;
    const __bf16* Kh = Kg + (size_t)n * T * D;
    const __bf16* Vh = Vg + (size_t)n * T * D;

    // This wave's 16 query rows as A-fragments for both D k-steps (D = 64).
    Frag qf[2];
    {
        const __bf16* qRow = Qh + (size_t)(qb * 128 + w * 16 + r) * D;
#pragma unroll
        for (int ks = 0; ks < 2; ++ks) {
            qf[ks].h[0] = *reinterpret_cast<const v8bf*>(qRow + ks * 32 + kh * 8);
            qf[ks].h[1] = *reinterpret_cast<const v8bf*>(qRow + ks * 32 + 16 + kh * 8);
        }
    }

    v8f accv[4];                           // ctx accumulators: 16 rows x 64 d
#pragma unroll
    for (int dt = 0; dt < 4; ++dt) accv[dt] = vzero8();
    float mst[8], lst[8];                  // row = e + 8*kh for accumulator slot e
#pragma unroll
    for (int e = 0; e < 8; ++e) { mst[e] = -1e30f; lst[e] = 0.0f; }

    for (int j = 0; j < T; j += 128) {
        __syncthreads();                   // previous tile fully consumed

#if HAVE_ASYNC_LDS
        // Async-stage K tile (128 keys x 64 d) row-major into padded LDS.
        // 1024 chunks of 16B; 4 per thread; per-lane LDS addresses keep pad.
#pragma unroll
        for (int it = 0; it < 4; ++it) {
            int idx = threadIdx.x + it * 256;
            int key = idx >> 3;                 // 0..127
            int c8  = (idx & 7) * 8;            // element offset within row
            __builtin_amdgcn_global_load_async_to_lds_b128(
                to_gbl_v4(Kh + (size_t)(j + key) * D + c8),
                to_lds_v4(&kT[key][c8]), 0, 0);
        }
#endif

        // Stage V tile (128 keys x 64 d) transposed into LDS (coalesced loads).
#pragma unroll
        for (int it = 0; it < 4; ++it) {
            int idx = threadIdx.x + it * 256;   // 1024 chunks of 8 elements
            int key = idx >> 3;
            int d0  = (idx & 7) * 8;
            v8bf tmp = *reinterpret_cast<const v8bf*>(Vh + (size_t)(j + key) * D + d0);
#pragma unroll
            for (int u = 0; u < 8; ++u) vT[d0 + u][key] = tmp[u];
        }
#if HAVE_ASYNC_LDS
        __builtin_amdgcn_s_wait_asynccnt(0);
#endif
        __syncthreads();

        // S(16x128) = Q @ K^T  (Q already scaled); K rows as B-fragments.
        v8f s[8];
#pragma unroll
        for (int nt = 0; nt < 8; ++nt) {
#if HAVE_ASYNC_LDS
            const __bf16* kRow = &kT[nt * 16 + r][0];
#else
            const __bf16* kRow = Kh + (size_t)(j + nt * 16 + r) * D;
#endif
            Frag kf0, kf1;
            kf0.h[0] = *reinterpret_cast<const v8bf*>(kRow + kh * 8);
            kf0.h[1] = *reinterpret_cast<const v8bf*>(kRow + 16 + kh * 8);
            kf1.h[0] = *reinterpret_cast<const v8bf*>(kRow + 32 + kh * 8);
            kf1.h[1] = *reinterpret_cast<const v8bf*>(kRow + 32 + 16 + kh * 8);
            s[nt] = __builtin_amdgcn_wmma_f32_16x16x32_bf16(
                false, qf[0].v, false, kf0.v, (short)0, vzero8(), false, false);
            s[nt] = __builtin_amdgcn_wmma_f32_16x16x32_bf16(
                false, qf[1].v, false, kf1.v, (short)0, s[nt], false, false);
        }

        // Online softmax; row reductions stay inside the 16-lane halves.
        float scale_old[8];
#pragma unroll
        for (int e = 0; e < 8; ++e) {
            float mx = s[0][e];
#pragma unroll
            for (int nt = 1; nt < 8; ++nt) mx = fmaxf(mx, s[nt][e]);
#pragma unroll
            for (int msk = 1; msk <= 8; msk <<= 1) mx = fmaxf(mx, __shfl_xor(mx, msk, 32));
            float mnew = fmaxf(mst[e], mx);
            float sc = __expf(mst[e] - mnew);
            float rowsum = 0.0f;
#pragma unroll
            for (int nt = 0; nt < 8; ++nt) {
                float p = __expf(s[nt][e] - mnew);
                s[nt][e] = p;
                rowsum += p;
            }
#pragma unroll
            for (int msk = 1; msk <= 8; msk <<= 1) rowsum += __shfl_xor(rowsum, msk, 32);
            lst[e] = lst[e] * sc + rowsum;
            mst[e] = mnew;
            scale_old[e] = sc;
        }
#pragma unroll
        for (int dt = 0; dt < 4; ++dt)
#pragma unroll
            for (int e = 0; e < 8; ++e) accv[dt][e] *= scale_old[e];

        // Repack P (bf16) into per-wave LDS row-major for A-fragment reads.
#pragma unroll
        for (int nt = 0; nt < 8; ++nt)
#pragma unroll
            for (int e = 0; e < 8; ++e)
                pS[w][e + 8 * kh][nt * 16 + r] = f2bf(s[nt][e]);
        // Same-wave LDS ops are in-order: no barrier needed before reading pS[w].

        // ctx += P @ V  (K-dim = 128 keys, 4 k-steps).
#pragma unroll
        for (int ks = 0; ks < 4; ++ks) {
            Frag pf;
            const __bf16* pRow = &pS[w][r][0];
            pf.h[0] = *reinterpret_cast<const v8bf*>(pRow + ks * 32 + kh * 8);
            pf.h[1] = *reinterpret_cast<const v8bf*>(pRow + ks * 32 + 16 + kh * 8);
#pragma unroll
            for (int dt = 0; dt < 4; ++dt) {
                Frag vf;
                const __bf16* vRow = &vT[dt * 16 + r][0];
                vf.h[0] = *reinterpret_cast<const v8bf*>(vRow + ks * 32 + kh * 8);
                vf.h[1] = *reinterpret_cast<const v8bf*>(vRow + ks * 32 + 16 + kh * 8);
                accv[dt] = __builtin_amdgcn_wmma_f32_16x16x32_bf16(
                    false, pf.v, false, vf.v, (short)0, accv[dt], false, false);
            }
        }
    }

    // Normalize and store ctx back in (T, B, E) layout (row m = t*B + b).
    const int bb = n >> 4, hh = n & 15;
#pragma unroll
    for (int dt = 0; dt < 4; ++dt)
#pragma unroll
        for (int e = 0; e < 8; ++e) {
            int tq  = qb * 128 + w * 16 + e + 8 * kh;
            int col = hh * 64 + dt * 16 + r;
            float val = accv[dt][e] / lst[e];
            Ctx[((size_t)tq * Bsz + bb) * E + col] = f2bf(val);
        }
}

// ---------------------------------------------------------------------------
// Output projection: Out = Ctx @ W2^T (fp32 output).
// ---------------------------------------------------------------------------
__global__ __launch_bounds__(256) void gemm_out(const __bf16* __restrict__ Ctx,
                                                const __bf16* __restrict__ W2,
                                                float* __restrict__ Out) {
    const int w    = threadIdx.x >> 5;
    const int lane = threadIdx.x & 31;
    const int wm = w & 1, wn = w >> 1;
    const int r = lane & 15, kh = lane >> 4;
    const int mBase = blockIdx.y * 128 + wm * 64;
    const int nBase = blockIdx.x * 128 + wn * 32;

    v8f acc[4][2];
    mm_wave_tile(Ctx, W2, mBase, nBase, acc);

#pragma unroll
    for (int i = 0; i < 4; ++i)
#pragma unroll
        for (int j = 0; j < 2; ++j)
#pragma unroll
            for (int e = 0; e < 8; ++e) {
                int m  = mBase + i * 16 + e + 8 * kh;
                int nn = nBase + j * 16 + r;
                Out[(size_t)m * E + nn] = acc[i][j][e];
            }
}

// ---------------------------------------------------------------------------
// Launcher
// ---------------------------------------------------------------------------
extern "C" void kernel_launch(void* const* d_in, const int* in_sizes, int n_in,
                              void* d_out, int out_size, void* d_ws, size_t ws_size,
                              hipStream_t stream) {
    (void)in_sizes; (void)n_in; (void)out_size; (void)ws_size;
    const float* q  = (const float*)d_in[0];   // (T, B, E) fp32
    const float* w1 = (const float*)d_in[1];   // (3E, E)  fp32
    const float* w2 = (const float*)d_in[2];   // (E, E)   fp32
    float* out = (float*)d_out;                // (T, B, E) fp32

    char* ws = (char*)d_ws;
    const size_t nX    = (size_t)M * E;          // 8M elems
    const size_t nW1   = (size_t)N_QKV * E;      // 3M elems
    const size_t nW2   = (size_t)E * E;          // 1M elems
    const size_t nHead = (size_t)Bsz * H * T * D;// 8M elems
    __bf16* Xbf = (__bf16*)ws;  ws += nX * 2;
    __bf16* W1b = (__bf16*)ws;  ws += nW1 * 2;
    __bf16* W2b = (__bf16*)ws;  ws += nW2 * 2;
    __bf16* Qb  = (__bf16*)ws;  ws += nHead * 2;
    __bf16* Kb  = (__bf16*)ws;  ws += nHead * 2;
    __bf16* Vb  = (__bf16*)ws;  ws += nHead * 2;
    __bf16* Ctx = (__bf16*)ws;  ws += nX * 2;

    cvt_f32_to_bf16<<<(int)((nX / 4 + 255) / 256), 256, 0, stream>>>(q,  Xbf, (int)(nX / 4));
    cvt_f32_to_bf16<<<(int)((nW1 / 4 + 255) / 256), 256, 0, stream>>>(w1, W1b, (int)(nW1 / 4));
    cvt_f32_to_bf16<<<(int)((nW2 / 4 + 255) / 256), 256, 0, stream>>>(w2, W2b, (int)(nW2 / 4));

    gemm_qkv<<<dim3(N_QKV / 128, M / 128), 256, 0, stream>>>(Xbf, W1b, Qb, Kb, Vb);
    flash_attn<<<Bsz * H * (T / 128), 256, 0, stream>>>(Qb, Kb, Vb, Ctx);
    gemm_out<<<dim3(E / 128, M / 128), 256, 0, stream>>>(Ctx, W2b, out);
}
